// QuantLinear_42700564857201
// MI455X (gfx1250) — matrix operand order repaired and checked by reading, not compiled, not run on hardware
//
#include <hip/hip_runtime.h>
#include <hip/hip_bf16.h>
#include <stdint.h>

#define M_DIM 16384
#define K_DIM 4096
#define N_DIM 4096
#define BM 128
#define BN 128
#define BK 32
#define LDP 40   // LDS row pitch in ushorts (32 data + 8 pad -> 80B = 20 banks, conflict-free b128 reads)

typedef __bf16 v16bf __attribute__((ext_vector_type(16)));
typedef __bf16 v8bf  __attribute__((ext_vector_type(8)));
typedef float  v8f   __attribute__((ext_vector_type(8)));

__device__ __forceinline__ unsigned short f2bf(float f) {
    unsigned u = __builtin_bit_cast(unsigned, f);
    unsigned r = 0x7FFFu + ((u >> 16) & 1u);
    return (unsigned short)((u + r) >> 16);
}

// inv[col_order[p]] = p
__global__ void build_inv_kernel(const int* __restrict__ col_order, int* __restrict__ inv) {
    int p = blockIdx.x * blockDim.x + threadIdx.x;
    if (p < K_DIM) inv[col_order[p]] = p;
}

// x fp32 -> bf16 (memory-bound one-pass; keeps conversion VALU out of the GEMM loop)
__global__ void convert_x_kernel(const float* __restrict__ x, unsigned short* __restrict__ xb) {
    const size_t i = ((size_t)blockIdx.x * blockDim.x + threadIdx.x) * 8;
    float4 a = *(const float4*)&x[i];
    float4 b = *(const float4*)&x[i + 4];
    uint4 o;
    o.x = (unsigned)f2bf(a.x) | ((unsigned)f2bf(a.y) << 16);
    o.y = (unsigned)f2bf(a.z) | ((unsigned)f2bf(a.w) << 16);
    o.z = (unsigned)f2bf(b.x) | ((unsigned)f2bf(b.y) << 16);
    o.w = (unsigned)f2bf(b.z) | ((unsigned)f2bf(b.w) << 16);
    *(uint4*)&xb[i] = o;
}

// Dequantize 4-bit weights (one value per int32) and store TRANSPOSED bf16: wT[n][k]
__global__ void dequant_kernel(const int* __restrict__ qweight,
                               const float* __restrict__ scales,
                               const int* __restrict__ qzeros,
                               const int* __restrict__ inv,
                               unsigned short* __restrict__ wT) {
    __shared__ unsigned short tile[64][72];
    const int tk = blockIdx.x * 64;
    const int tn = blockIdx.y * 64;
    const int t  = threadIdx.x;
    {
        const int nl = t & 63;
        const int ks = (t >> 6) * 16;
        const int n  = tn + nl;
        #pragma unroll 4
        for (int r = 0; r < 16; ++r) {
            const int kl = ks + r;
            const int k  = tk + kl;
            const int g  = inv[k] >> 7;
            const int q  = qweight[(size_t)k * N_DIM + n];
            const int z  = (qzeros[g * (N_DIM / 8) + (n >> 3)] >> ((n & 7) * 4)) & 0xF;
            const float s = scales[g * N_DIM + n];
            tile[kl][nl] = f2bf((float)(q - z) * s);
        }
    }
    __syncthreads();
    {
        const int klw = t & 63;
        const int nsw = (t >> 6) * 16;
        #pragma unroll 4
        for (int r = 0; r < 16; ++r) {
            const int nlw = nsw + r;
            wT[(size_t)(tn + nlw) * K_DIM + (tk + klw)] = tile[klw][nlw];
        }
    }
}

__global__ __launch_bounds__(256)
void gemm_kernel(const unsigned short* __restrict__ xb,
                 const unsigned short* __restrict__ wT,
                 const float* __restrict__ bias,
                 float* __restrict__ out) {
    __shared__ unsigned short As[2][BM * LDP];
    __shared__ unsigned short Bs[2][BN * LDP];

    const int tid   = threadIdx.x;
    const int lane  = tid & 31;
    const int wid   = tid >> 5;
    const int wm    = wid >> 2;          // 0..1
    const int wn    = wid & 3;           // 0..3
    const int mBase = blockIdx.y * BM;
    const int nBase = blockIdx.x * BN;
    const int lhalf = (lane < 16) ? 0 : 1;
    const int l16   = lane & 15;

    const v8f zero = {};
    v8f c[4][2];
    #pragma unroll
    for (int i = 0; i < 4; ++i)
        #pragma unroll
        for (int j = 0; j < 2; ++j) c[i][j] = zero;

    // Async-DMA staging: each tile = 128 rows x 4 b128-chunks; 2 chunks/thread for A and for B.
    unsigned long long aG[2], bG[2];   // per-thread global source addresses (incremental)
    unsigned aL[2], bL[2];             // LDS dest offsets within one buffer
    #pragma unroll
    for (int i = 0; i < 2; ++i) {
        const int slot = tid + i * 256;
        const int row  = slot >> 2, ch = slot & 3;
        aG[i] = (unsigned long long)(uintptr_t)xb + 2ull * ((size_t)(mBase + row) * K_DIM + ch * 8);
        bG[i] = (unsigned long long)(uintptr_t)wT + 2ull * ((size_t)(nBase + row) * K_DIM + ch * 8);
        aL[i] = (unsigned)(row * LDP * 2 + ch * 16);
        bL[i] = (unsigned)(row * LDP * 2 + ch * 16);
    }
    const unsigned asBase    = (unsigned)(uintptr_t)&As[0][0];
    const unsigned bsBase    = (unsigned)(uintptr_t)&Bs[0][0];
    const unsigned bufStride = (unsigned)(BM * LDP * 2);   // 10240 B, same for A and B tiles

    // Issue the 4 async b128 copies for the next K-tile into buffer 'dstSel', then advance.
    auto issueAsync = [&](unsigned dstSel) {
        #pragma unroll
        for (int i = 0; i < 2; ++i) {
            asm volatile("global_load_async_to_lds_b128 %0, %1, off"
                         :: "v"(asBase + dstSel + aL[i]), "v"(aG[i]) : "memory");
        }
        #pragma unroll
        for (int i = 0; i < 2; ++i) {
            asm volatile("global_load_async_to_lds_b128 %0, %1, off"
                         :: "v"(bsBase + dstSel + bL[i]), "v"(bG[i]) : "memory");
        }
        #pragma unroll
        for (int i = 0; i < 2; ++i) { aG[i] += BK * 2; bG[i] += BK * 2; }
    };

    const int akb = lhalf * 8;   // A frag: lanes<16 -> K0..7 & K16..23 ; lanes>=16 -> K8..15 & K24..31
    const int bkb = lhalf * 8;   // B frag: lanes<16 -> K0..7 & K16..23 ; lanes>=16 -> K8..15 & K24..31
    const char* aFragBase = (const char*)&As[0][0] + (wm * 64 + l16) * (LDP * 2) + akb * 2;
    const char* bFragBase = (const char*)&Bs[0][0] + (wn * 32 + l16) * (LDP * 2) + bkb * 2;

    auto compute = [&](unsigned sel) {
        const char* A = aFragBase + sel;
        const char* B = bFragBase + sel;
        v16bf afrag[4];
        v16bf bfrag[2];
        #pragma unroll
        for (int i = 0; i < 4; ++i) {
            v8bf lo = *(const v8bf*)(A + i * 16 * (LDP * 2));
            v8bf hi = *(const v8bf*)(A + i * 16 * (LDP * 2) + 32);
            afrag[i] = __builtin_shufflevector(lo, hi, 0,1,2,3,4,5,6,7,8,9,10,11,12,13,14,15);
        }
        #pragma unroll
        for (int j = 0; j < 2; ++j) {
            v8bf lo = *(const v8bf*)(B + j * 16 * (LDP * 2));
            v8bf hi = *(const v8bf*)(B + j * 16 * (LDP * 2) + 32);
            bfrag[j] = __builtin_shufflevector(lo, hi, 0,1,2,3,4,5,6,7,8,9,10,11,12,13,14,15);
        }
        #pragma unroll
        for (int i = 0; i < 4; ++i)
            #pragma unroll
            for (int j = 0; j < 2; ++j)
                c[i][j] = __builtin_amdgcn_wmma_f32_16x16x32_bf16(
                    false, afrag[i], false, bfrag[j], (short)0, c[i][j], false, false);
    };

    const int NK = K_DIM / BK;   // 128
    unsigned sel = 0;
    issueAsync(0);               // prime buffer 0 with K-tile 0
    // Steady state: branch-free body; always issue DMA for tile kt+1 into the idle buffer.
    for (int kt = 0; kt < NK - 1; ++kt) {
        asm volatile("s_wait_asynccnt 0x0" ::: "memory");   // own DMAs for 'sel' done
        __syncthreads();                                     // everyone's DMAs done + prior readers done
        issueAsync(sel ^ bufStride);                         // overlaps with WMMA below
        compute(sel);
        sel ^= bufStride;
    }
    asm volatile("s_wait_asynccnt 0x0" ::: "memory");
    __syncthreads();
    compute(sel);

    // Epilogue: C layout -> VGPR v: M = v (lanes 0-15) / v+8 (lanes 16-31), N = lane%16
    #pragma unroll
    for (int i = 0; i < 4; ++i) {
        #pragma unroll
        for (int j = 0; j < 2; ++j) {
            const int col = nBase + wn * 32 + j * 16 + l16;
            const float bv = bias[col];
            const int rbase = mBase + wm * 64 + i * 16 + lhalf * 8;
            #pragma unroll
            for (int v = 0; v < 8; ++v) {
                out[(size_t)(rbase + v) * N_DIM + col] = c[i][j][v] + bv;
            }
        }
    }
}

extern "C" void kernel_launch(void* const* d_in, const int* in_sizes, int n_in,
                              void* d_out, int out_size, void* d_ws, size_t ws_size,
                              hipStream_t stream) {
    const float* x         = (const float*)d_in[0];
    const int*   qweight   = (const int*)d_in[1];
    const float* scales    = (const float*)d_in[2];
    const int*   qzeros    = (const int*)d_in[3];
    const int*   col_order = (const int*)d_in[4];
    const float* bias      = (const float*)d_in[5];
    float*       out       = (float*)d_out;

    // Workspace layout: inv (16 KB) | wT bf16 K*N (32 MB) | xb bf16 M*K (128 MB)
    int* inv = (int*)d_ws;
    unsigned short* wT = (unsigned short*)((char*)d_ws + 16384);
    unsigned short* xb = (unsigned short*)((char*)d_ws + 16384 + (size_t)K_DIM * N_DIM * 2);

    build_inv_kernel<<<K_DIM / 256, 256, 0, stream>>>(col_order, inv);

    dim3 gDq(K_DIM / 64, N_DIM / 64);
    dequant_kernel<<<gDq, 256, 0, stream>>>(qweight, scales, qzeros, inv, wT);

    const size_t xElems = (size_t)M_DIM * K_DIM;
    convert_x_kernel<<<(unsigned)(xElems / 8 / 256), 256, 0, stream>>>(x, xb);

    dim3 gGemm(N_DIM / BN, M_DIM / BM);
    gemm_kernel<<<gGemm, 256, 0, stream>>>(xb, wT, bias, out);
}
